// Recommender_80590766342898
// MI455X (gfx1250) — matrix operand order repaired and checked by reading, not compiled, not run on hardware
//
#include <hip/hip_runtime.h>
#include <hip/hip_bf16.h>

// ---------------------------------------------------------------------------
// MI455X (gfx1250, wave32) implementation.
// All GEMMs run through v_wmma_f32_16x16x32_bf16 (bf16 inputs, f32 accum).
// Gathers (the memory-bound part) stay f32/float4 coalesced, accumulate f32.
// ---------------------------------------------------------------------------

typedef __bf16 bf16;
typedef __attribute__((ext_vector_type(16))) __bf16 v16bf;
typedef __attribute__((ext_vector_type(8)))  __bf16 v8bf;
typedef __attribute__((ext_vector_type(8)))  float  v8f;

#define DEV static __device__ __forceinline__

// A fragment (16x32 bf16, M x K) from a row-major LDS tile.
// Contiguous k order used here (same as load_b_frag): lanes 0-15 -> M=lane, K in {kh*16+0..15}, kh=lane>>4.
DEV v16bf load_a_frag_lds(const bf16* A, int lda, int k0, int lane) {
  const int m  = lane & 15;
  const int kh = lane >> 4;
  const bf16* p = A + m * lda + k0 + kh * 16;
  v8bf lo = *(const v8bf*)(p);
  v8bf hi = *(const v8bf*)(p + 8);
  v16bf r;
#pragma unroll
  for (int i = 0; i < 8; ++i) { r[i] = lo[i]; r[8 + i] = hi[i]; }
  return r;
}

// B fragment (32x16 bf16, K x N) where B[k,n] = W[n0+n, k0+k], W row-major [N,Kfull].
// Contiguous k order used here (same as A): lanes 0-15 hold K=0..15, lanes 16-31 hold K=16..31, N=lane&15.
// 32B-aligned contiguous 16-bf16 run per lane -> single v16bf load.
DEV v16bf load_b_frag(const bf16* W, int ldk, int n0, int k0, int lane) {
  const int n  = lane & 15;
  const int kh = lane >> 4;
  const bf16* p = W + (size_t)(n0 + n) * ldk + k0 + kh * 16;
  return *(const v16bf*)p;
}

DEV v8f wmma_bf16(v16bf a, v16bf b, v8f c) {
  // (neg_a, A, neg_b, B, c_mod, C, reuse_a, reuse_b)
  return __builtin_amdgcn_wmma_f32_16x16x32_bf16(false, a, false, b, (short)0, c,
                                                 false, false);
}

// C/D layout: c[i] holds row m = i (+8 for upper half-wave), col n = lane&15.
DEV int cd_row(int i, int lane) { return (lane < 16) ? i : (i + 8); }

DEV float elu_f(float x) { return (x > 0.f) ? x : expm1f(x); }

// ---------------------------------------------------------------------------
// Weight f32 -> bf16 conversion
// ---------------------------------------------------------------------------
__global__ void f32_to_bf16_kernel(const float* __restrict__ src,
                                   bf16* __restrict__ dst, int n) {
  int i = blockIdx.x * 256 + threadIdx.x;
  if (i < n) dst[i] = (bf16)src[i];
}

// ---------------------------------------------------------------------------
// News encoder: out = tanh(elu(title @ W1^T + b1) @ W2^T + b2)
// 1 block = 16 news rows, 8 waves = 8 column tiles of 16.
// ---------------------------------------------------------------------------
__global__ __launch_bounds__(256)
void news_encode_kernel(const float* __restrict__ titles,
                        const int* __restrict__ cand_news,
                        const int* __restrict__ clicked_news,
                        const bf16* __restrict__ W1, const float* __restrict__ b1,
                        const bf16* __restrict__ W2, const float* __restrict__ b2,
                        bf16* __restrict__ out_e) {
  __shared__ bf16 sT[16 * 768];   // staged title tile (bf16)
  __shared__ bf16 sH[16 * 128];   // hidden activations
  __shared__ int  sId[16];
  const int tid  = threadIdx.x;
  const int lane = tid & 31;
  const int wave = tid >> 5;
  const int row0 = blockIdx.x * 16;

  if (tid < 16) {
    int r = row0 + tid;                       // combined order: [cand | clicked]
    sId[tid] = (r < 320) ? cand_news[r] : clicked_news[r - 320];
  }
  __syncthreads();
  for (int idx = tid; idx < 16 * 768; idx += 256) {
    int r = idx / 768, c = idx - r * 768;
    sT[idx] = (bf16)titles[(size_t)sId[r] * 768 + c];
  }
  __syncthreads();

  const int n0 = wave * 16;
  v8f acc = {};
#pragma unroll 4
  for (int k0 = 0; k0 < 768; k0 += 32) {
    __builtin_prefetch(W1 + (size_t)n0 * 768 + k0 + 64);  // stream next weight tile
    v16bf a = load_a_frag_lds(sT, 768, k0, lane);
    v16bf b = load_b_frag(W1, 768, n0, k0, lane);
    acc = wmma_bf16(a, b, acc);
  }
#pragma unroll
  for (int i = 0; i < 8; ++i) {
    int m = cd_row(i, lane), n = lane & 15;
    sH[m * 128 + n0 + n] = (bf16)elu_f(acc[i] + b1[n0 + n]);
  }
  __syncthreads();

  v8f acc2 = {};
#pragma unroll
  for (int k0 = 0; k0 < 128; k0 += 32) {
    v16bf a = load_a_frag_lds(sH, 128, k0, lane);
    v16bf b = load_b_frag(W2, 128, n0, k0, lane);
    acc2 = wmma_bf16(a, b, acc2);
  }
#pragma unroll
  for (int i = 0; i < 8; ++i) {
    int m = cd_row(i, lane), n = lane & 15;
    out_e[(size_t)(row0 + m) * 128 + n0 + n] = (bf16)tanhf(acc2[i] + b2[n0 + n]);
  }
}

// ---------------------------------------------------------------------------
// Anchor graph attention: 1 block per news row (20 anchors, padded to 32 rows).
// gather+agg (f32, coalesced float4 rows) -> h = tanh(X@Wa^T+ba) (WMMA)
// -> u = elu(h@W1^T+b1) (WMMA) -> logits = u@w2+b2 -> softmax -> sum(w*h).
// ---------------------------------------------------------------------------
__global__ __launch_bounds__(256)
void anchor_kernel(const float* __restrict__ ent, const float* __restrict__ rel,
                   const int* __restrict__ cand_anchor, const int* __restrict__ clk_anchor,
                   const int* __restrict__ ent_adj, const int* __restrict__ rel_adj,
                   const bf16* __restrict__ Wa, const float* __restrict__ ba,
                   const bf16* __restrict__ W1, const float* __restrict__ b1,
                   const float* __restrict__ W2, const float* __restrict__ b2,
                   bf16* __restrict__ out_e) {
  __shared__ bf16  sX[32 * 256];   // [node_e | agg] bf16, rows 20..31 zero pad
  __shared__ bf16  sH[32 * 128];
  __shared__ float sLogit[32];
  __shared__ float sW[32];
  const int tid  = threadIdx.x;
  const int lane = tid & 31;
  const int wave = tid >> 5;
  const int nrow = blockIdx.x;
  const int* nodes = (nrow < 320) ? (cand_anchor + (size_t)nrow * 20)
                                  : (clk_anchor + (size_t)(nrow - 320) * 20);

  for (int idx = tid; idx < 12 * 256; idx += 256) sX[20 * 256 + idx] = (bf16)0.f;
  if (tid < 32) sLogit[tid] = 0.f;

  // gather + aggregate: one wave per anchor, 16B (float4) per lane per row
  for (int a = wave; a < 20; a += 8) {
    int node = nodes[a];
    float4 ne = ((const float4*)(ent + (size_t)node * 128))[lane];
    float4 agg = make_float4(0.f, 0.f, 0.f, 0.f);
#pragma unroll
    for (int k = 0; k < 10; ++k) {
      int e = ent_adj[node * 10 + k];
      int r = rel_adj[node * 10 + k];
      float4 ve = ((const float4*)(ent + (size_t)e * 128))[lane];
      float4 vr = ((const float4*)(rel + (size_t)r * 128))[lane];
      agg.x += ve.x + vr.x; agg.y += ve.y + vr.y;
      agg.z += ve.z + vr.z; agg.w += ve.w + vr.w;
    }
    const int d0 = lane * 4;
    bf16* pn = &sX[a * 256 + d0];
    pn[0] = (bf16)ne.x;  pn[1] = (bf16)ne.y;  pn[2] = (bf16)ne.z;  pn[3] = (bf16)ne.w;
    bf16* pg = &sX[a * 256 + 128 + d0];
    pg[0] = (bf16)agg.x; pg[1] = (bf16)agg.y; pg[2] = (bf16)agg.z; pg[3] = (bf16)agg.w;
  }
  __syncthreads();

  const int n0 = wave * 16;
  // h = tanh(X @ Wa^T + ba), M=32 (2 m-tiles per wave), K=256
#pragma unroll
  for (int mt = 0; mt < 2; ++mt) {
    v8f acc = {};
#pragma unroll
    for (int k0 = 0; k0 < 256; k0 += 32) {
      v16bf a = load_a_frag_lds(sX + mt * 16 * 256, 256, k0, lane);
      v16bf b = load_b_frag(Wa, 256, n0, k0, lane);
      acc = wmma_bf16(a, b, acc);
    }
#pragma unroll
    for (int i = 0; i < 8; ++i) {
      int m = mt * 16 + cd_row(i, lane), n = lane & 15;
      sH[m * 128 + n0 + n] = (bf16)tanhf(acc[i] + ba[n0 + n]);
    }
  }
  __syncthreads();

  // u = elu(h @ W1^T + b1); logits[m] += u * w2[n]
#pragma unroll
  for (int mt = 0; mt < 2; ++mt) {
    v8f acc = {};
#pragma unroll
    for (int k0 = 0; k0 < 128; k0 += 32) {
      v16bf a = load_a_frag_lds(sH + mt * 16 * 128, 128, k0, lane);
      v16bf b = load_b_frag(W1, 128, n0, k0, lane);
      acc = wmma_bf16(a, b, acc);
    }
#pragma unroll
    for (int i = 0; i < 8; ++i) {
      int m = mt * 16 + cd_row(i, lane), n = lane & 15;
      float u = elu_f(acc[i] + b1[n0 + n]);
      atomicAdd(&sLogit[m], u * W2[n0 + n]);   // ds_add_f32
    }
  }
  __syncthreads();

  if (tid == 0) {
    const float bias = b2[0];
    float mx = -3.4e38f;
    for (int a = 0; a < 20; ++a) mx = fmaxf(mx, sLogit[a] + bias);
    float s = 0.f;
    for (int a = 0; a < 20; ++a) { float e = __expf(sLogit[a] + bias - mx); sW[a] = e; s += e; }
    float inv = 1.f / s;
    for (int a = 0; a < 20; ++a) sW[a] *= inv;
  }
  __syncthreads();

  if (tid < 128) {
    float s = 0.f;
    for (int a = 0; a < 20; ++a) s += sW[a] * (float)sH[a * 128 + tid];
    out_e[(size_t)nrow * 128 + tid] = (bf16)s;
  }
}

// ---------------------------------------------------------------------------
// Final MLP: out = elu(elu([news_e|anchor_e] @ W1^T + b1) @ W2^T + b2), f32 out
// ---------------------------------------------------------------------------
__global__ __launch_bounds__(256)
void mlp_kernel(const bf16* __restrict__ news_e, const bf16* __restrict__ anchor_e,
                const bf16* __restrict__ W1, const float* __restrict__ b1,
                const bf16* __restrict__ W2, const float* __restrict__ b2,
                float* __restrict__ out_final) {
  __shared__ bf16 sX[16 * 256];
  __shared__ bf16 sY[16 * 128];
  const int tid  = threadIdx.x;
  const int lane = tid & 31;
  const int wave = tid >> 5;
  const int row0 = blockIdx.x * 16;

  for (int idx = tid; idx < 16 * 256; idx += 256) {
    int r = idx >> 8, c = idx & 255;
    sX[idx] = (c < 128) ? news_e[(size_t)(row0 + r) * 128 + c]
                        : anchor_e[(size_t)(row0 + r) * 128 + (c - 128)];
  }
  __syncthreads();

  const int n0 = wave * 16;
  v8f acc = {};
#pragma unroll
  for (int k0 = 0; k0 < 256; k0 += 32) {
    v16bf a = load_a_frag_lds(sX, 256, k0, lane);
    v16bf b = load_b_frag(W1, 256, n0, k0, lane);
    acc = wmma_bf16(a, b, acc);
  }
#pragma unroll
  for (int i = 0; i < 8; ++i) {
    int m = cd_row(i, lane), n = lane & 15;
    sY[m * 128 + n0 + n] = (bf16)elu_f(acc[i] + b1[n0 + n]);
  }
  __syncthreads();

  v8f acc2 = {};
#pragma unroll
  for (int k0 = 0; k0 < 128; k0 += 32) {
    v16bf a = load_a_frag_lds(sY, 128, k0, lane);
    v16bf b = load_b_frag(W2, 128, n0, k0, lane);
    acc2 = wmma_bf16(a, b, acc2);
  }
#pragma unroll
  for (int i = 0; i < 8; ++i) {
    int m = cd_row(i, lane), n = lane & 15;
    out_final[(size_t)(row0 + m) * 128 + n0 + n] = elu_f(acc2[i] + b2[n0 + n]);
  }
}

// ---------------------------------------------------------------------------
// Score: user = mean over clicked; score[b,s] = dot(user, cand[s])
// ---------------------------------------------------------------------------
__global__ __launch_bounds__(128)
void score_kernel(const float* __restrict__ finalv, float* __restrict__ out) {
  __shared__ float sU[128];
  __shared__ float sR[128];
  const int b = blockIdx.x, t = threadIdx.x;
  float u = 0.f;
  for (int c = 0; c < 50; ++c)
    u += finalv[(size_t)(320 + b * 50 + c) * 128 + t];
  sU[t] = u * (1.f / 50.f);
  __syncthreads();
  for (int s = 0; s < 5; ++s) {
    sR[t] = sU[t] * finalv[(size_t)(b * 5 + s) * 128 + t];
    __syncthreads();
    for (int off = 64; off > 0; off >>= 1) {
      if (t < off) sR[t] += sR[t + off];
      __syncthreads();
    }
    if (t == 0) out[b * 5 + s] = sR[0];
    __syncthreads();
  }
}

// ---------------------------------------------------------------------------
extern "C" void kernel_launch(void* const* d_in, const int* in_sizes, int n_in,
                              void* d_out, int out_size, void* d_ws, size_t ws_size,
                              hipStream_t stream) {
  (void)in_sizes; (void)n_in; (void)out_size; (void)ws_size;
  const float* titles    = (const float*)d_in[0];
  const float* ent       = (const float*)d_in[1];
  const float* rel       = (const float*)d_in[2];
  const float* W_nc1     = (const float*)d_in[3];
  const float* b_nc1     = (const float*)d_in[4];
  const float* W_nc2     = (const float*)d_in[5];
  const float* b_nc2     = (const float*)d_in[6];
  const float* W_anchor  = (const float*)d_in[7];
  const float* b_anchor  = (const float*)d_in[8];
  const float* W_aw1     = (const float*)d_in[9];
  const float* b_aw1     = (const float*)d_in[10];
  const float* W_aw2     = (const float*)d_in[11];
  const float* b_aw2     = (const float*)d_in[12];
  const float* W_mlp1    = (const float*)d_in[13];
  const float* b_mlp1    = (const float*)d_in[14];
  const float* W_mlp2    = (const float*)d_in[15];
  const float* b_mlp2    = (const float*)d_in[16];
  const int* cand_news   = (const int*)d_in[17];
  const int* clicked_news= (const int*)d_in[18];
  const int* cand_anchor = (const int*)d_in[19];
  const int* clk_anchor  = (const int*)d_in[20];
  const int* ent_adj     = (const int*)d_in[21];
  const int* rel_adj     = (const int*)d_in[22];

  char* ws = (char*)d_ws;
  size_t off = 0;
  auto wsAlloc = [&](size_t bytes) {
    char* p = ws + off;
    off = (off + bytes + 255) & ~(size_t)255;
    return p;
  };
  bf16* bW_nc1    = (bf16*)wsAlloc((size_t)128 * 768 * 2);
  bf16* bW_nc2    = (bf16*)wsAlloc((size_t)128 * 128 * 2);
  bf16* bW_anchor = (bf16*)wsAlloc((size_t)128 * 256 * 2);
  bf16* bW_aw1    = (bf16*)wsAlloc((size_t)128 * 128 * 2);
  bf16* bW_mlp1   = (bf16*)wsAlloc((size_t)128 * 256 * 2);
  bf16* bW_mlp2   = (bf16*)wsAlloc((size_t)128 * 128 * 2);
  bf16* news_e    = (bf16*)wsAlloc((size_t)3520 * 128 * 2);
  bf16* anchor_e  = (bf16*)wsAlloc((size_t)3520 * 128 * 2);
  float* finalv   = (float*)wsAlloc((size_t)3520 * 128 * 4);

  auto conv = [&](const float* s, bf16* d, int n) {
    f32_to_bf16_kernel<<<(n + 255) / 256, 256, 0, stream>>>(s, d, n);
  };
  conv(W_nc1,    bW_nc1,    128 * 768);
  conv(W_nc2,    bW_nc2,    128 * 128);
  conv(W_anchor, bW_anchor, 128 * 256);
  conv(W_aw1,    bW_aw1,    128 * 128);
  conv(W_mlp1,   bW_mlp1,   128 * 256);
  conv(W_mlp2,   bW_mlp2,   128 * 128);

  // 3520 combined news rows = 220 tiles of 16
  news_encode_kernel<<<220, 256, 0, stream>>>(titles, cand_news, clicked_news,
                                              bW_nc1, b_nc1, bW_nc2, b_nc2, news_e);
  anchor_kernel<<<3520, 256, 0, stream>>>(ent, rel, cand_anchor, clk_anchor,
                                          ent_adj, rel_adj, bW_anchor, b_anchor,
                                          bW_aw1, b_aw1, W_aw2, b_aw2, anchor_e);
  mlp_kernel<<<220, 256, 0, stream>>>(news_e, anchor_e, bW_mlp1, b_mlp1,
                                      bW_mlp2, b_mlp2, finalv);
  score_kernel<<<64, 128, 0, stream>>>(finalv, (float*)d_out);
}
